// SentenceRNN_16870631539331
// MI455X (gfx1250) — matrix operand order repaired and checked by reading, not compiled, not run on hardware
//
#include <hip/hip_runtime.h>
#include <hip/hip_bf16.h>
#include <cstdint>
#include <cstddef>

// ---------------------------------------------------------------------------
// Problem constants (from reference): B=256 S=6 R=64 D=1024 H=1024 A=2*H=2048
// ---------------------------------------------------------------------------
#define B_ 256
#define S_ 6
#define R_ 64
#define D_ 1024
#define H_ 1024
#define A_ 2048

typedef __attribute__((ext_vector_type(16))) __bf16 v16bf;
typedef __attribute__((ext_vector_type(8)))  float  v8f;
typedef __attribute__((ext_vector_type(4)))  __bf16 v4bf;

// ---------------------------------------------------------------------------
// fp32 -> bf16 conversion (vectorized, 4 elems/thread)
// ---------------------------------------------------------------------------
__global__ void cvt_f32_bf16(const float* __restrict__ src,
                             __bf16* __restrict__ dst, long n4) {
  long i = (long)blockIdx.x * blockDim.x + threadIdx.x;
  if (i >= n4) return;
  float4 f = ((const float4*)src)[i];
  v4bf o;
  o.x = (__bf16)f.x; o.y = (__bf16)f.y; o.z = (__bf16)f.z; o.w = (__bf16)f.w;
  ((v4bf*)dst)[i] = o;
}

// ---------------------------------------------------------------------------
// copy initial h/c/topic states
// ---------------------------------------------------------------------------
__global__ void init_state(const float* __restrict__ h0, const float* __restrict__ c0,
                           const float* __restrict__ t0,
                           float* __restrict__ hS, float* __restrict__ cS,
                           float* __restrict__ tS) {
  int idx = blockIdx.x * 256 + threadIdx.x;           // B*H
  hS[idx] = h0[idx]; cS[idx] = c0[idx]; tS[idx] = t0[idx];
}

// ---------------------------------------------------------------------------
// Generic WMMA GEMM:  C[M x N] = A[M x K] (row major, bf16)
//                               @ W[N x K]^T (row major, bf16)  (+ bias [+bias2])
// Block: 256 threads = 8 wave32, tile 128(M) x 128(N), BK=32.
// Wave (wm,wn in 4x2): 32x64 patch = 2x4 tiles of 16x16 via
// v_wmma_f32_16x16x32_bf16, fp32 accumulate.
//
// Tiles are staged global->LDS with CDNA5 async-DMA
// (GLOBAL_LOAD_ASYNC_TO_LDS_B128, ASYNCcnt-tracked) into a double buffer:
// the fill of tile k+1 runs behind the WMMAs of tile k. Each thread issues
// 2 async b128s for A and 2 for B per tile (INST_OFFSET adds to both the LDS
// and global address per the ISA, so the two 16B chunks share operands).
//
// Row addressing supports grouped rows: row r -> (r/rpg)*groupStride + (r%rpg)*lda
// (used so att1's A-operand can read x in its native (B,S,R,D) layout).
// flags: 1 = bf16 output, 2 = accumulate into C, 4 = fused sigmoid
// ---------------------------------------------------------------------------
#define ASYNC_LD32(ldsAddr, gPtr)                                             \
  asm volatile("global_load_async_to_lds_b128 %0, %1, off\n\t"                \
               "global_load_async_to_lds_b128 %0, %1, off offset:16"          \
               :: "v"(ldsAddr), "v"(gPtr) : "memory")

__global__ __launch_bounds__(256)
void gemm_wmma_bf16(const __bf16* __restrict__ Abase, int lda, int rpg, long groupStride,
                    const __bf16* __restrict__ W, int ldw,
                    void* __restrict__ Cout, int ldc,
                    const float* __restrict__ bias, const float* __restrict__ bias2,
                    int K, int flags) {
  __shared__ __align__(16) __bf16 As[2][128][32];
  __shared__ __align__(16) __bf16 Bs[2][128][32];

  const int tid  = threadIdx.x;
  const int lane = tid & 31;
  const int wave = tid >> 5;
  const int wm   = wave >> 1;        // 0..3
  const int wn   = wave & 1;         // 0..1
  const int hi   = lane >> 4;        // half-wave select
  const int ln   = lane & 15;

  const long rowBlock = (long)blockIdx.y * 128;
  const long colBlock = (long)blockIdx.x * 128;

  v8f acc[2][4];
  for (int mt = 0; mt < 2; ++mt)
    for (int nt = 0; nt < 4; ++nt)
      for (int j = 0; j < 8; ++j) acc[mt][nt][j] = 0.f;

  // cooperative tile fill: each thread owns one 32B row-chunk per tile
  const int ldRow   = tid >> 1;            // 0..127
  const int ldChunk = (tid & 1) * 16;      // 0 or 16 (elements)
  const long aRowG  = rowBlock + ldRow;
  const __bf16* aPtr = Abase + (aRowG / rpg) * groupStride
                             + (aRowG % rpg) * (long)lda + ldChunk;
  const __bf16* wPtr = W + (colBlock + ldRow) * (long)ldw + ldChunk;

  // LDS byte addresses for this thread's chunk in each buffer
  // (low 32 bits of the flat shared address == LDS offset per aperture rules)
  const uint32_t aLds0 = (uint32_t)(uintptr_t)&As[0][ldRow][ldChunk];
  const uint32_t aLds1 = (uint32_t)(uintptr_t)&As[1][ldRow][ldChunk];
  const uint32_t bLds0 = (uint32_t)(uintptr_t)&Bs[0][ldRow][ldChunk];
  const uint32_t bLds1 = (uint32_t)(uintptr_t)&Bs[1][ldRow][ldChunk];

  const int nIter = K >> 5;                // K / 32
  // prologue: start DMA of tile 0 into buffer 0
  ASYNC_LD32(aLds0, aPtr);
  ASYNC_LD32(bLds0, wPtr);

  for (int it = 0; it < nIter; ++it) {
    const int cur = it & 1;
    if (it + 1 < nIter) {
      const __bf16* aN = aPtr + (long)(it + 1) * 32;
      const __bf16* wN = wPtr + (long)(it + 1) * 32;
      if (cur == 0) { ASYNC_LD32(aLds1, aN); ASYNC_LD32(bLds1, wN); }
      else          { ASYNC_LD32(aLds0, aN); ASYNC_LD32(bLds0, wN); }
      // 4 asyncs/wave per buffer: <=4 outstanding => current tile landed
      asm volatile("s_wait_asynccnt 0x4" ::: "memory");
    } else {
      asm volatile("s_wait_asynccnt 0x0" ::: "memory");
    }
    __syncthreads();   // all waves' DMA for `cur` visible

    // A fragment (16-bit 16x32): lane L<16 holds row M=L, K in [hi*8,hi*8+8)
    // and [16+hi*8, 24+hi*8); lane L+16 holds the complementary K octets.
    union { v16bf v; uint4 u[2]; } af[2], bfr[4];
    for (int mt = 0; mt < 2; ++mt) {
      int row = wm * 32 + mt * 16 + ln;
      af[mt].u[0] = *(const uint4*)&As[cur][row][hi * 8];
      af[mt].u[1] = *(const uint4*)&As[cur][row][16 + hi * 8];
    }
    // B fragment (16-bit 32x16): lane holds N=ln, K in [hi*8,hi*8+8) and [16+hi*8,24+hi*8).
    for (int nt = 0; nt < 4; ++nt) {
      int col = wn * 64 + nt * 16 + ln;
      bfr[nt].u[0] = *(const uint4*)&Bs[cur][col][hi * 8];
      bfr[nt].u[1] = *(const uint4*)&Bs[cur][col][16 + hi * 8];
    }
    for (int mt = 0; mt < 2; ++mt)
      for (int nt = 0; nt < 4; ++nt)
        acc[mt][nt] = __builtin_amdgcn_wmma_f32_16x16x32_bf16(
            false, af[mt].v, false, bfr[nt].v, (short)0, acc[mt][nt], false, false);
    __syncthreads();   // all waves done reading `cur` before it is refilled
  }

  // Epilogue. C layout: VGPR j -> M = j + hi*8, N = ln.
  const bool outBf16 = flags & 1;
  const bool doAcc   = flags & 2;
  const bool doSig   = flags & 4;
  float*  Cf = (float*)Cout;
  __bf16* Cb = (__bf16*)Cout;
  for (int mt = 0; mt < 2; ++mt)
    for (int nt = 0; nt < 4; ++nt) {
      long row0 = rowBlock + wm * 32 + mt * 16 + hi * 8;
      long col  = colBlock + wn * 64 + nt * 16 + ln;
      float badd = 0.f;
      if (bias)  badd += bias[col];
      if (bias2) badd += bias2[col];
      for (int j = 0; j < 8; ++j) {
        long off = (row0 + j) * (long)ldc + col;
        float v = acc[mt][nt][j] + badd;
        if (doAcc) v += Cf[off];
        if (doSig) v = 1.f / (1.f + __expf(-v));
        if (outBf16) Cb[off] = (__bf16)v; else Cf[off] = v;
      }
    }
}

// ---------------------------------------------------------------------------
// Attention energy + softmax. One block per batch element b.
// e[b,r] = sum_a tanh(att1[b,r,a] + att2[b,a]) * W_full[a]   (b_full dropped:
// softmax is shift-invariant). Wave w reduces rows r = w*8..w*8+7.
// ---------------------------------------------------------------------------
__global__ __launch_bounds__(256)
void attn_e_softmax(const __bf16* __restrict__ att1, const float* __restrict__ att2,
                    const float* __restrict__ wfull, const int* __restrict__ lengths,
                    float* __restrict__ alpha, float* __restrict__ outAlpha, int step) {
  __shared__ float satt2[A_];
  __shared__ float swf[A_];
  __shared__ float se[R_];
  __shared__ float sExp[R_];
  __shared__ float sSum;
  const int b   = blockIdx.x;
  const int tid = threadIdx.x;
  for (int a = tid; a < A_; a += 256) {
    satt2[a] = att2[(long)b * A_ + a];
    swf[a]   = wfull[a];
  }
  __syncthreads();

  const int lane = tid & 31, wave = tid >> 5;
  const __bf16* base = att1 + (long)b * R_ * A_;
  for (int rr = 0; rr < 8; ++rr) {
    int r = wave * 8 + rr;
    const __bf16* p = base + (long)r * A_;
    float partial = 0.f;
    for (int a = lane; a < A_; a += 32)
      partial += tanhf((float)p[a] + satt2[a]) * swf[a];
    for (int off = 16; off; off >>= 1) partial += __shfl_xor(partial, off, 32);
    if (lane == 0) se[r] = partial;
  }
  __syncthreads();

  if (tid == 0) {
    float mx = se[0];
    for (int r = 1; r < R_; ++r) mx = fmaxf(mx, se[r]);
    float sum = 0.f;
    for (int r = 0; r < R_; ++r) { float e = __expf(se[r] - mx); sExp[r] = e; sum += e; }
    sSum = sum;
  }
  __syncthreads();

  if (tid < R_) {
    float al = sExp[tid] / sSum;
    alpha[(long)b * R_ + tid] = al;
    bool m = step < lengths[b];
    outAlpha[((long)b * S_ + step) * R_ + tid] = m ? al : 0.f;
  }
}

// attw[b,d] = sum_r x[b,step,r,d] * alpha[b,r]
__global__ void attw_kernel(const __bf16* __restrict__ xb, const float* __restrict__ alpha,
                            float* __restrict__ attw, int step) {
  int idx = blockIdx.x * 256 + threadIdx.x;       // B*D
  int b = idx >> 10, d = idx & (D_ - 1);
  const __bf16* xp = xb + (((long)b * S_ + step) * R_) * D_ + d;
  const float*  al = alpha + (long)b * R_;
  float s = 0.f;
  for (int r = 0; r < R_; ++r) s += (float)xp[(long)r * D_] * al[r];
  attw[idx] = s;
}

// tcb[b] = concat(topic[b], gate[b]*attw[b]) as bf16
__global__ void ctx_kernel(const float* __restrict__ topicS, const float* __restrict__ gate,
                           const float* __restrict__ attw, __bf16* __restrict__ tcb) {
  int idx = blockIdx.x * 256 + threadIdx.x;       // B*H
  int b = idx >> 10, j = idx & (H_ - 1);
  tcb[(long)b * 2 * H_ + j]      = (__bf16)topicS[idx];
  tcb[(long)b * 2 * H_ + H_ + j] = (__bf16)(gate[idx] * attw[idx]);
}

// LSTM pointwise update + length masking; writes hiddens_all slice.
__global__ void lstm_kernel(const float* __restrict__ gates, float* __restrict__ hS,
                            float* __restrict__ cS, float* __restrict__ tS,
                            const int* __restrict__ lengths, float* __restrict__ outH,
                            int step) {
  int idx = blockIdx.x * 256 + threadIdx.x;       // B*H
  int b = idx >> 10, j = idx & (H_ - 1);
  long gb = (long)b * 4 * H_;
  float ig = gates[gb + j], fg = gates[gb + H_ + j];
  float gg = gates[gb + 2 * H_ + j], og = gates[gb + 3 * H_ + j];
  float si = 1.f / (1.f + __expf(-ig));
  float sf = 1.f / (1.f + __expf(-fg));
  float so = 1.f / (1.f + __expf(-og));
  float cn = sf * cS[idx] + si * tanhf(gg);
  float hn = so * tanhf(cn);
  bool m = step < lengths[b];
  if (m) { hS[idx] = hn; cS[idx] = cn; tS[idx] = hn; }
  outH[((long)b * S_ + step) * H_ + j] = m ? hn : 0.f;
}

// ---------------------------------------------------------------------------
// Host driver
// ---------------------------------------------------------------------------
extern "C" void kernel_launch(void* const* d_in, const int* in_sizes, int n_in,
                              void* d_out, int out_size, void* d_ws, size_t ws_size,
                              hipStream_t stream) {
  const float* x      = (const float*)d_in[0];
  const float* t0     = (const float*)d_in[1];
  const float* h0     = (const float*)d_in[2];
  const float* c0     = (const float*)d_in[3];
  const float* Wenc   = (const float*)d_in[4];
  const float* benc   = (const float*)d_in[5];
  const float* Wdec   = (const float*)d_in[6];
  const float* bdec   = (const float*)d_in[7];
  const float* Wfull  = (const float*)d_in[8];
  /* d_in[9] = b_full: unused (softmax shift-invariance) */
  const float* Wfbeta = (const float*)d_in[10];
  const float* bfbeta = (const float*)d_in[11];
  const float* Wih    = (const float*)d_in[12];
  const float* Whh    = (const float*)d_in[13];
  const float* bih    = (const float*)d_in[14];
  const float* bhh    = (const float*)d_in[15];
  const int*   lens   = (const int*)d_in[16];

  float* outH = (float*)d_out;                       // (B,S,H)
  float* outA = (float*)d_out + (long)B_ * S_ * H_;  // (B,S,R)

  // ---- workspace carve-up (256B aligned) ----
  char* wsp = (char*)d_ws;
  auto carve = [&](size_t bytes) {
    char* p = wsp; wsp += (bytes + 255) & ~(size_t)255; return (void*)p;
  };
  const size_t nX = (size_t)B_ * S_ * R_ * D_;
  __bf16* xb      = (__bf16*)carve(nX * 2);
  __bf16* wencb   = (__bf16*)carve((size_t)A_ * D_ * 2);
  __bf16* wdecb   = (__bf16*)carve((size_t)A_ * H_ * 2);
  __bf16* wfbetab = (__bf16*)carve((size_t)H_ * H_ * 2);
  __bf16* wihb    = (__bf16*)carve((size_t)4 * H_ * 2 * H_ * 2);
  __bf16* whhb    = (__bf16*)carve((size_t)4 * H_ * H_ * 2);
  __bf16* att1    = (__bf16*)carve((size_t)B_ * R_ * A_ * 2);   // per-step
  float*  att2    = (float*)carve((size_t)B_ * A_ * 4);
  float*  alpha   = (float*)carve((size_t)B_ * R_ * 4);
  float*  attw    = (float*)carve((size_t)B_ * H_ * 4);
  float*  gate    = (float*)carve((size_t)B_ * H_ * 4);
  float*  gates   = (float*)carve((size_t)B_ * 4 * H_ * 4);
  float*  hS      = (float*)carve((size_t)B_ * H_ * 4);
  float*  cS      = (float*)carve((size_t)B_ * H_ * 4);
  float*  tS      = (float*)carve((size_t)B_ * H_ * 4);
  __bf16* hb      = (__bf16*)carve((size_t)B_ * H_ * 2);
  __bf16* tcb     = (__bf16*)carve((size_t)B_ * 2 * H_ * 2);

  auto cvt = [&](const float* s, __bf16* d, size_t n) {
    long n4 = (long)(n / 4);
    cvt_f32_bf16<<<(unsigned)((n4 + 255) / 256), 256, 0, stream>>>(s, d, n4);
  };
  // one-time (per launch) bf16 casts
  cvt(x, xb, nX);
  cvt(Wenc, wencb, (size_t)A_ * D_);
  cvt(Wdec, wdecb, (size_t)A_ * H_);
  cvt(Wfbeta, wfbetab, (size_t)H_ * H_);
  cvt(Wih, wihb, (size_t)4 * H_ * 2 * H_);
  cvt(Whh, whhb, (size_t)4 * H_ * H_);
  init_state<<<B_ * H_ / 256, 256, 0, stream>>>(h0, c0, t0, hS, cS, tS);

  const int BIG = 1 << 30;   // rowsPerGroup for ungrouped operands
  for (int s = 0; s < S_; ++s) {
    // h (fp32 state) -> bf16 operand for this step's GEMMs
    cvt(hS, hb, (size_t)B_ * H_);

    // att2 = h @ W_dec^T + b_dec          [B x A]
    gemm_wmma_bf16<<<dim3(A_ / 128, B_ / 128), 256, 0, stream>>>(
        hb, H_, BIG, 0, wdecb, H_, att2, A_, bdec, nullptr, H_, 0);

    // att1 = x_s @ W_enc^T + b_enc        [(B*R) x A], bf16 out, grouped rows
    gemm_wmma_bf16<<<dim3(A_ / 128, (B_ * R_) / 128), 256, 0, stream>>>(
        xb + (long)s * R_ * D_, D_, R_, (long)S_ * R_ * D_,
        wencb, D_, att1, A_, benc, nullptr, D_, /*bf16 out*/1);

    // e + softmax -> alpha (and masked alphas output)
    attn_e_softmax<<<B_, 256, 0, stream>>>(att1, att2, Wfull, lens, alpha, outA, s);

    // attw = sum_r alpha * x_s
    attw_kernel<<<B_ * D_ / 256, 256, 0, stream>>>(xb, alpha, attw, s);

    // gate = sigmoid(h @ W_fbeta^T + b_fbeta)
    gemm_wmma_bf16<<<dim3(H_ / 128, B_ / 128), 256, 0, stream>>>(
        hb, H_, BIG, 0, wfbetab, H_, gate, H_, bfbeta, nullptr, H_, /*sigmoid*/4);

    // tcb = concat(topic, gate*attw) as bf16
    ctx_kernel<<<B_ * H_ / 256, 256, 0, stream>>>(tS, gate, attw, tcb);

    // gates = tcb @ W_ih^T + b_ih + b_hh ; then += h @ W_hh^T
    gemm_wmma_bf16<<<dim3(4 * H_ / 128, B_ / 128), 256, 0, stream>>>(
        tcb, 2 * H_, BIG, 0, wihb, 2 * H_, gates, 4 * H_, bih, bhh, 2 * H_, 0);
    gemm_wmma_bf16<<<dim3(4 * H_ / 128, B_ / 128), 256, 0, stream>>>(
        hb, H_, BIG, 0, whhb, H_, gates, 4 * H_, nullptr, nullptr, H_, /*acc*/2);

    // LSTM pointwise + masked state update + hiddens output
    lstm_kernel<<<B_ * H_ / 256, 256, 0, stream>>>(gates, hS, cS, tS, lens, outH, s);
  }
}
